// GPTOSSMoeMLP2Block_27547920237057
// MI455X (gfx1250) — compile-verified, matrix operand order checked by reading, not run on hardware
//
#include <hip/hip_runtime.h>

// Problem dims (fixed by the reference)
#define B_TOK 16
#define K_SL  4
#define NPAIR 64     // B*K routed (token, slot) pairs
#define NEXP  64
#define HD    2880
#define ID    2880

// Tiling
#define TILE_H 64            // H rows per workgroup (16 per wave)
#define WAVES  4
#define CHUNK  960           // I-dimension chunk staged in LDS (3 chunks of 960)
#define NCHUNK 3
#define LDS_STRIDE (CHUNK + 8)  // pad: 968 elems -> 484 dwords/row, 484%64=36 -> conflict-free lane map

typedef __attribute__((ext_vector_type(16))) __bf16 v16bf;
typedef __attribute__((ext_vector_type(8)))  __bf16 v8bf;
typedef __attribute__((ext_vector_type(8)))  float  v8f;

// ---------------------------------------------------------------------------
// Kernel 1: deterministic bucketing of the 64 (b,k) pairs by expert index.
// Single thread: 64 iterations, negligible cost, bit-stable pair ordering.
// ---------------------------------------------------------------------------
__global__ void moe_bucket_kernel(const int* __restrict__ idx,
                                  int* __restrict__ cnts,
                                  int* __restrict__ pairs) {
  if (threadIdx.x == 0 && blockIdx.x == 0) {
    int c[NEXP];
    #pragma unroll
    for (int e = 0; e < NEXP; ++e) c[e] = 0;
    for (int p = 0; p < NPAIR; ++p) {
      int e = idx[p] & (NEXP - 1);
      pairs[e * NPAIR + c[e]] = p;
      c[e]++;
    }
    #pragma unroll
    for (int e = 0; e < NEXP; ++e) cnts[e] = c[e];
  }
}

// ---------------------------------------------------------------------------
// Kernel 2: per-expert grouped GEMM via WMMA.
// grid = (HD/TILE_H, NEXP); block = 128 (4 wave32).
// Each wave: D[16h x 16tok] += A(W tile 16x32 bf16) x B(act tile 32x16 bf16)
// A streamed from HBM (bandwidth path), B staged in LDS, C/D fp32.
// Epilogue: acc[pair*HD + h] = gate * (dot + bias)   (each slot written once)
// ---------------------------------------------------------------------------
__global__ __launch_bounds__(WAVES * 32) void moe_mlp2_kernel(
    const __bf16* __restrict__ act,    // [NPAIR][ID]
    const __bf16* __restrict__ gates,  // [NPAIR]
    const int*    __restrict__ cnts,   // [NEXP]
    const int*    __restrict__ pairs,  // [NEXP][NPAIR]
    const __bf16* __restrict__ W,      // [NEXP][HD][ID]
    const __bf16* __restrict__ bias,   // [NEXP][HD]
    float*        __restrict__ acc)    // [NPAIR][HD]
{
  const int e   = blockIdx.y;
  const int cnt = cnts[e];
  if (cnt == 0) return;  // expert not routed: no weight traffic at all

  const int tid    = threadIdx.x;
  const int lane   = tid & 31;
  const int wave   = tid >> 5;
  const int h0w    = blockIdx.x * TILE_H + wave * 16;  // this wave's 16 H rows
  const int rowsel = lane & 15;                        // A row / B column for this lane
  const int koff   = (lane < 16) ? 0 : 8;              // ISA k-split: lanes>=16 own k 8..15 / 24..31

  __shared__ __bf16 actT[16][LDS_STRIDE];

  const int ngroups = (cnt + 15) >> 4;   // >16 tokens on one expert: loop column groups
  for (int g = 0; g < ngroups; ++g) {
    v8f cacc = {};  // fp32 C/D accumulator, 16x16 tile

    for (int ch = 0; ch < NCHUNK; ++ch) {
      // -------- stage up to 16 activation columns (this I-chunk) into LDS ----
      const int chunks16 = CHUNK * 2 / 16;  // 16B chunks per row = 120
      for (int t = tid; t < 16 * chunks16; t += blockDim.x) {
        const int m   = t / chunks16;   // column slot 0..15
        const int o16 = t % chunks16;
        uint4 v = make_uint4(0u, 0u, 0u, 0u);
        const int gm = g * 16 + m;
        if (gm < cnt) {
          const int pr = pairs[e * NPAIR + gm];
          v = *(const uint4*)(act + (size_t)pr * ID + (size_t)ch * CHUNK + (size_t)o16 * 8);
        }
        *(uint4*)(&actT[m][o16 * 8]) = v;  // zero-fill unused columns
      }
      __syncthreads();

      const __bf16* arow = W + ((size_t)e * HD + h0w + rowsel) * (size_t)ID
                             + (size_t)ch * CHUNK + koff;
      const __bf16* brow = &actT[rowsel][koff];

      #pragma unroll 5
      for (int i0 = 0; i0 < CHUNK; i0 += 32) {
        __builtin_prefetch(arow + i0 + CHUNK);          // global_prefetch_b8: stream next chunk
        v8bf alo = *(const v8bf*)(arow + i0);           // k = i0 + koff .. +7
        v8bf ahi = *(const v8bf*)(arow + i0 + 16);      // k = i0 + 16 + koff .. +7
        v8bf blo = *(const v8bf*)(brow + i0);
        v8bf bhi = *(const v8bf*)(brow + i0 + 16);
        v16bf A, Bm;
        #pragma unroll
        for (int j = 0; j < 8; ++j) {
          A[j] = alo[j];  A[j + 8] = ahi[j];
          Bm[j] = blo[j]; Bm[j + 8] = bhi[j];
        }
        // D = A(16x32 bf16) x B(32x16 bf16) + C(f32)
        cacc = __builtin_amdgcn_wmma_f32_16x16x32_bf16(
            /*neg_a=*/false, A, /*neg_b=*/false, Bm,
            /*c_mod=*/(short)0, cacc, /*reuse_a=*/false, /*reuse_b=*/false);
      }
      __syncthreads();
    }

    // -------- epilogue: bias + gate, one exclusive writer per (pair, h) -----
    // D layout: lane holds column N = lane&15; VGPR v holds row M = v + (lane<16?0:8)
    const int m = g * 16 + (lane & 15);
    if (m < cnt) {
      const int   pr    = pairs[e * NPAIR + m];
      const float gate  = (float)gates[pr];
      const int   hbase = h0w + koff;
      #pragma unroll
      for (int v = 0; v < 8; ++v) {
        const int   h  = hbase + v;
        const float bv = (float)bias[(size_t)e * HD + h];
        acc[(size_t)pr * HD + h] = gate * (cacc[v] + bv);
      }
    }
    __syncthreads();
  }
}

// ---------------------------------------------------------------------------
// Kernel 3: combine the K=4 slot contributions per token, fp32 -> bf16.
// ---------------------------------------------------------------------------
__global__ void moe_combine_kernel(const float* __restrict__ acc,
                                   __bf16* __restrict__ out) {
  const int t = blockIdx.x * blockDim.x + threadIdx.x;
  if (t >= B_TOK * HD) return;
  const int b = t / HD;
  const int h = t - b * HD;
  float s = 0.f;
  #pragma unroll
  for (int k = 0; k < K_SL; ++k)
    s += acc[(size_t)(b * K_SL + k) * HD + h];
  out[t] = (__bf16)s;
}

// ---------------------------------------------------------------------------
extern "C" void kernel_launch(void* const* d_in, const int* in_sizes, int n_in,
                              void* d_out, int out_size, void* d_ws, size_t ws_size,
                              hipStream_t stream) {
  (void)in_sizes; (void)n_in; (void)out_size; (void)ws_size;

  const __bf16* act   = (const __bf16*)d_in[0];  // [B,K,I] bf16
  const __bf16* gates = (const __bf16*)d_in[1];  // [B,K]   bf16
  const int*    idx   = (const int*)   d_in[2];  // [B,K]   i32
  const __bf16* W     = (const __bf16*)d_in[3];  // [E,H,I] bf16
  const __bf16* bias  = (const __bf16*)d_in[4];  // [E,H]   bf16
  __bf16*       out   = (__bf16*)d_out;          // [B,H]   bf16

  // workspace layout: [NPAIR*HD f32 acc][NEXP i32 cnts][NEXP*NPAIR i32 pairs]
  float* acc  = (float*)d_ws;
  int*   cnts = (int*)((char*)d_ws + (size_t)NPAIR * HD * sizeof(float));
  int*   prs  = cnts + NEXP;

  moe_bucket_kernel<<<1, 32, 0, stream>>>(idx, cnts, prs);

  dim3 grid(HD / TILE_H, NEXP);
  moe_mlp2_kernel<<<grid, WAVES * 32, 0, stream>>>(act, gates, cnts, prs, W, bias, acc);

  const int total = B_TOK * HD;
  moe_combine_kernel<<<(total + 255) / 256, 256, 0, stream>>>(acc, out);
}
